// SupConLoss_54520314856040
// MI455X (gfx1250) — matrix operand order repaired and checked by reading, not compiled, not run on hardware
//
#include <hip/hip_runtime.h>
#include <hip/hip_bf16.h>

typedef __bf16 bf16;
typedef __attribute__((ext_vector_type(16))) __bf16 v16bf;
typedef __attribute__((ext_vector_type(8)))  __bf16 v8bf;
typedef __attribute__((ext_vector_type(8)))  float  v8f;
typedef __attribute__((ext_vector_type(4)))  float  v4f;

#define SB    4096              // batch size B
#define SN    8192              // N = B * V
#define SD    256               // feature dim
#define NCLS  1000              // number of classes
#define LDSROW 264              // 256 + 8 bf16 pad -> conflict-free ds_load_b128
#define K2E   2.8853900817779268f   // 2 * log2(e)

// exp2 on the TRANS pipe; compile-safe fallback
#if __has_builtin(__builtin_amdgcn_exp2f)
#define EXP2F(x) __builtin_amdgcn_exp2f(x)
#else
#define EXP2F(x) exp2f(x)
#endif

// ---- workspace layout (byte offsets, all 256-aligned) ----
#define WS_ACC   0                              // 1 float  : loss accumulator
#define WS_CSUM  256                            // 1000*256 f32 : per-class anchor sums
#define WS_HIST  (WS_CSUM + NCLS * SD * 4)      // 1000 int : class histogram (x2 views)
#define WS_S     (WS_HIST + 4096)               // 8192 f32 : per-row exp-sum
#define WS_DIAG  (WS_S + SN * 4)                // 8192 f32 : per-row self-dot
#define WS_ANCH  (WS_DIAG + SN * 4)             // 8192*256 bf16 : normalized anchors

// ---------------------------------------------------------------------------
// Kernel 0: zero accumulator + class sums + histogram
// ---------------------------------------------------------------------------
__global__ void supcon_zero(char* ws) {
    int i = blockIdx.x * 256 + threadIdx.x;
    if (i < WS_S / 4) ((float*)ws)[i] = 0.0f;
}

// ---------------------------------------------------------------------------
// Kernel 1: normalize rows -> bf16 anchors; also build class-sum table,
// class histogram, and per-row bf16 self-dot (the exact GEMM diagonal).
// One wave per row n = v*B + b; anchor[n] = feats[b][v] / max(||.||,1e-12).
// ---------------------------------------------------------------------------
__global__ void __launch_bounds__(256)
supcon_prep(const float* __restrict__ feats, const int* __restrict__ labels,
            char* __restrict__ ws) {
    bf16*  anchor = (bf16*)(ws + WS_ANCH);
    float* csum   = (float*)(ws + WS_CSUM);
    int*   hist   = (int*)(ws + WS_HIST);
    float* diag   = (float*)(ws + WS_DIAG);

    const int wave = threadIdx.x >> 5;
    const int lane = threadIdx.x & 31;
    const int n    = blockIdx.x * 8 + wave;
    const int v    = n >> 12;
    const int b    = n & (SB - 1);
    const float* src = feats + ((size_t)b * 2 + v) * SD;

    v4f x0 = *(const v4f*)(src + lane * 8);
    v4f x1 = *(const v4f*)(src + lane * 8 + 4);

    float ss = 0.0f;
#pragma unroll
    for (int i = 0; i < 4; ++i) ss += x0[i] * x0[i] + x1[i] * x1[i];
#pragma unroll
    for (int m = 1; m <= 16; m <<= 1) ss += __shfl_xor(ss, m, 32);

    const float s = 1.0f / fmaxf(sqrtf(ss), 1e-12f);

    v8bf o;
    float q[8];
#pragma unroll
    for (int i = 0; i < 4; ++i) {
        o[i]     = (bf16)(x0[i] * s);
        o[4 + i] = (bf16)(x1[i] * s);
    }
#pragma unroll
    for (int i = 0; i < 8; ++i) q[i] = (float)o[i];
    *(v8bf*)(anchor + (size_t)n * SD + lane * 8) = o;

    // bf16-quantized self-dot (matches the GEMM diagonal)
    float sd = 0.0f;
#pragma unroll
    for (int i = 0; i < 8; ++i) sd += q[i] * q[i];
#pragma unroll
    for (int m = 1; m <= 16; m <<= 1) sd += __shfl_xor(sd, m, 32);
    if (lane == 0) diag[n] = sd;

    // class-sum table + histogram (over all N rows -> counts include both views)
    const int lab = labels[b];
    float* cr = csum + (size_t)lab * SD + lane * 8;
#pragma unroll
    for (int i = 0; i < 8; ++i) atomicAdd(cr + i, q[i]);
    if (lane == 0) atomicAdd(hist + lab, 1);
}

// ---------------------------------------------------------------------------
// Kernel 2: S[n] = sum_m exp(2*<a_n,a_m> - 2)   (softmax denom, shift = 2.0)
// 4 waves/block, 16 rows/wave (A frags resident in 64 VGPRs). Column tiles
// staged 4-at-a-time (32 KB) through padded LDS and shared by all 4 waves:
// 4x less L2 traffic. Inner loop: v_wmma_f32_16x16x32_bf16 + fma/exp2/add.
// ---------------------------------------------------------------------------
__global__ void __launch_bounds__(128)
supcon_main(char* __restrict__ ws) {
    const bf16* anchor = (const bf16*)(ws + WS_ANCH);
    float*      Sarr   = (float*)(ws + WS_S);

    __shared__ bf16 lds[64 * LDSROW];           // 4 tiles x 16 rows, padded

    const int tid  = threadIdx.x;
    const int wave = tid >> 5;
    const int lane = tid & 31;
    const int half = lane >> 4;
    const int l15  = lane & 15;
    const int rowBase = blockIdx.x * 64 + wave * 16;

    // A fragments: rows rowBase..+15, all K. Contiguous k order (same as B): lane holds
    // M = lane%16; elems 0..15 at K = kc*32 + 16*half + s.
    const bf16* arow = anchor + (size_t)(rowBase + l15) * SD;
    v16bf A[8];
#pragma unroll
    for (int kc = 0; kc < 8; ++kc) {
        v8bf lo = *(const v8bf*)(arow + kc * 32 + half * 16);
        v8bf hi = *(const v8bf*)(arow + kc * 32 + half * 16 + 8);
        A[kc] = __builtin_shufflevector(lo, hi, 0, 1, 2, 3, 4, 5, 6, 7,
                                                8, 9, 10, 11, 12, 13, 14, 15);
    }

    float S[8];
#pragma unroll
    for (int r = 0; r < 8; ++r) S[r] = 0.0f;

    // staging assignment: 2 threads per column-row, 256 B each
    const int  srow = tid >> 1, shalf = tid & 1;
    const bf16* gstage = anchor + (size_t)srow * SD + shalf * 128;
    bf16*       lstage = lds + srow * LDSROW + shalf * 128;

    for (int round = 0; round < SN / 64; ++round) {
        const bf16* g = gstage + (size_t)round * 64 * SD;
#pragma unroll
        for (int i = 0; i < 16; ++i)
            *(v8bf*)(lstage + i * 8) = *(const v8bf*)(g + i * 8);
        if (round + 1 < SN / 64)
            __builtin_prefetch(g + 64 * SD, 0, 1);   // -> global_prefetch_b8
        __syncthreads();

#pragma unroll
        for (int tt = 0; tt < 4; ++tt) {
            // B layout: lane holds N = lane%16; one 32 B chunk at
            // K = kc*32 + 16*half. LDS row stride 528 B -> conflict-free.
            const bf16* bb = lds + (tt * 16 + l15) * LDSROW + half * 16;
            v8f acc = {};
#pragma unroll
            for (int kc = 0; kc < 8; ++kc) {
                v16bf B = *(const v16bf*)(bb + kc * 32);
                acc = __builtin_amdgcn_wmma_f32_16x16x32_bf16(
                    false, A[kc], false, B, (short)0, acc, false, false);
            }
#pragma unroll
            for (int r = 0; r < 8; ++r)
                S[r] += EXP2F(fmaf(acc[r], K2E, -K2E));   // exp(2d-2)
        }
        __syncthreads();
    }

    // reduce each row's S across the 16 lanes of its half, write per-row
#pragma unroll
    for (int r = 0; r < 8; ++r) {
#pragma unroll
        for (int m = 1; m <= 8; m <<= 1) S[r] += __shfl_xor(S[r], m, 32);
    }
    if (l15 == 0) {
#pragma unroll
        for (int r = 0; r < 8; ++r)
            Sarr[rowBase + r + half * 8] = S[r];
    }
}

// ---------------------------------------------------------------------------
// Kernel 3: per-row finalize. P[n] = 2*<a_n, csum[lab]>, npos = hist[lab]-1,
// denom = S[n] - exp(l_nn - 2), log_prob = (P - l_nn - 2*npos) - npos*log(denom)
// ---------------------------------------------------------------------------
__global__ void __launch_bounds__(256)
supcon_fin(const int* __restrict__ labels, char* __restrict__ ws) {
    const bf16*  anchor = (const bf16*)(ws + WS_ANCH);
    const float* csum   = (const float*)(ws + WS_CSUM);
    const int*   hist   = (const int*)(ws + WS_HIST);
    const float* Sarr   = (const float*)(ws + WS_S);
    const float* diag   = (const float*)(ws + WS_DIAG);
    float*       acc    = (float*)(ws + WS_ACC);

    const int wave = threadIdx.x >> 5;
    const int lane = threadIdx.x & 31;
    const int n    = blockIdx.x * 8 + wave;
    const int lab  = labels[n & (SB - 1)];

    v8bf av = *(const v8bf*)(anchor + (size_t)n * SD + lane * 8);
    const float* cr = csum + (size_t)lab * SD + lane * 8;
    v4f c0 = *(const v4f*)(cr);
    v4f c1 = *(const v4f*)(cr + 4);

    float p = 0.0f;
#pragma unroll
    for (int i = 0; i < 4; ++i)
        p += (float)av[i] * c0[i] + (float)av[4 + i] * c1[i];
#pragma unroll
    for (int m = 1; m <= 16; m <<= 1) p += __shfl_xor(p, m, 32);

    if (lane == 0) {
        float P     = 2.0f * p;                    // sum of logits over same-class (incl self)
        float lnn   = 2.0f * diag[n];              // diagonal logit
        float denom = Sarr[n] - __expf(lnn - 2.0f);
        float npos  = (float)hist[lab] - 1.0f;     // exclude self
        float logp  = (P - lnn - npos * 2.0f) - npos * __logf(denom);
        atomicAdd(acc, (npos > 0.0f) ? logp / npos : 0.0f);
    }
}

// ---------------------------------------------------------------------------
// Kernel 4: loss = -mean(log_probs)
// ---------------------------------------------------------------------------
__global__ void supcon_final(const char* ws, float* out) {
    if (threadIdx.x == 0) out[0] = -((const float*)(ws + WS_ACC))[0] / (float)SN;
}

// ---------------------------------------------------------------------------
extern "C" void kernel_launch(void* const* d_in, const int* in_sizes, int n_in,
                              void* d_out, int out_size, void* d_ws, size_t ws_size,
                              hipStream_t stream) {
    const float* features = (const float*)d_in[0];   // [4096, 2, 256] f32
    const int*   labels   = (const int*)d_in[1];     // [4096]
    float*       out      = (float*)d_out;           // scalar loss
    char*        ws       = (char*)d_ws;

    supcon_zero <<<(WS_S / 4 + 255) / 256, 256, 0, stream>>>(ws);
    supcon_prep <<<SN / 8,  256, 0, stream>>>(features, labels, ws);
    supcon_main <<<SN / 64, 128, 0, stream>>>(ws);
    supcon_fin  <<<SN / 8,  256, 0, stream>>>(labels, ws);
    supcon_final<<<1, 1, 0, stream>>>(ws, out);
}
